// HyperGraphStructuralLayer_GN_19825569038845
// MI455X (gfx1250) — matrix operand order repaired and checked by reading, not compiled, not run on hardware
//
#include <hip/hip_runtime.h>
#include <hip/hip_bf16.h>

#define NN 50000      // nodes
#define NE 50000      // hyperedges
#define EC 320000     // incidences
#define DD 256        // feature dim

typedef __bf16 bf16_t;
typedef bf16_t bf16x4  __attribute__((ext_vector_type(4)));
typedef bf16_t bf16x8  __attribute__((ext_vector_type(8)));
typedef bf16_t bf16x16 __attribute__((ext_vector_type(16)));
typedef float  floatx8 __attribute__((ext_vector_type(8)));
typedef int    intx4   __attribute__((vector_size(16)));

union BF16x16U { bf16x16 v; bf16x8 h[2]; };

#if defined(__has_builtin)
#if __has_builtin(__builtin_amdgcn_global_load_async_to_lds_b128) && \
    __has_builtin(__builtin_amdgcn_s_wait_asynccnt)
#define USE_ASYNC_LDS 1
#endif
#endif
#ifndef USE_ASYNC_LDS
#define USE_ASYNC_LDS 0
#endif

#if USE_ASYNC_LDS
typedef __attribute__((address_space(1))) intx4* gptr_t;   // prints as "int __vector(4) __device__ *"
typedef __attribute__((address_space(3))) intx4* lptr_t;   // prints as "int __vector(4) __shared__ *"
#endif

// ---------------- utility kernels ----------------

__global__ __launch_bounds__(256) void zero4_kernel(float4* __restrict__ p, int n4) {
    int i = blockIdx.x * blockDim.x + threadIdx.x;
    if (i < n4) p[i] = make_float4(0.f, 0.f, 0.f, 0.f);
}

__global__ __launch_bounds__(256) void cvt_f32_to_bf16x4_kernel(const float4* __restrict__ in,
                                                                bf16x4* __restrict__ out, int n4) {
    int i = blockIdx.x * blockDim.x + threadIdx.x;
    if (i < n4) {
        float4 v = in[i];
        bf16x4 o;
        o.x = (bf16_t)v.x; o.y = (bf16_t)v.y; o.z = (bf16_t)v.z; o.w = (bf16_t)v.w;
        out[i] = o;
    }
}

__global__ __launch_bounds__(256) void count_deg_kernel(const int* __restrict__ src,
                                                        const int* __restrict__ he,
                                                        float* __restrict__ deg_n,
                                                        float* __restrict__ deg_e, int e) {
    int i = blockIdx.x * blockDim.x + threadIdx.x;
    if (i < e) {
        atomicAdd(deg_n + src[i], 1.0f);
        atomicAdd(deg_e + he[i], 1.0f);
    }
}

__global__ __launch_bounds__(256) void invert_kernel(float* __restrict__ a, float* __restrict__ b, int n) {
    int i = blockIdx.x * blockDim.x + threadIdx.x;
    if (i < n) {
        float va = a[i]; a[i] = (va > 0.f) ? (1.0f / va) : 0.0f;
        float vb = b[i]; b[i] = (vb > 0.f) ? (1.0f / vb) : 0.0f;
    }
}

// ---------------- WMMA bf16 GEMM:  C[M,256] = A[M,256] x W[d,k]^T ----------------
// One wave computes a 16-row strip of A held entirely in registers (8 x v16bf frags).
// W (the WMMA B operand; row-major W == column-major B) is staged into LDS in two
// 64KB halves via the CDNA5 async global->LDS path, then fed to WMMA via ds loads.

__global__ __launch_bounds__(256) void gemm_bf16_wmma_kernel(const bf16_t* __restrict__ A,
                                                             const bf16_t* __restrict__ W,
                                                             float* __restrict__ C, int M) {
    __shared__ bf16_t Wlds[128 * DD];            // 64 KB: half of W (128 output cols)

    const int wave  = threadIdx.x >> 5;
    const int lane  = threadIdx.x & 31;
    const int strip = blockIdx.x * 8 + wave;
    const bool active = (strip * 16 < M);        // wave-uniform
    const int m0   = strip * 16;
    const int lrow = lane & 15;
    const int lhi  = lane >> 4;                  // 0 or 1

    // Load full 16x256 A strip as 8 fragments of 16x32 (contiguous k = 16*lhi + s order, same as B)
    bf16x16 afrag[8];
    if (active) {
        const bf16_t* arow = A + (size_t)(m0 + lrow) * DD;
        #pragma unroll
        for (int ks = 0; ks < 8; ++ks) {
            const bf16_t* p0 = arow + ks * 32 + lhi * 16;  // K 0..7 / 16..23
            const bf16_t* p1 = p0 + 8;                     // K 8..15 / 24..31
            BF16x16U u;
            u.h[0] = *(const bf16x8*)p0;
            u.h[1] = *(const bf16x8*)p1;
            afrag[ks] = u.v;
        }
    }

    #pragma unroll 1
    for (int half = 0; half < 2; ++half) {
        // ---- stage 64KB of W into LDS (all 256 threads cooperate) ----
        const char* wsrc = (const char*)(W + (size_t)half * 128 * DD);
#if USE_ASYNC_LDS
        #pragma unroll
        for (int j = 0; j < 16; ++j) {
            int byteoff = j * 4096 + threadIdx.x * 16;     // lane-contiguous 16B chunks
            __builtin_amdgcn_global_load_async_to_lds_b128(
                (gptr_t)(wsrc + byteoff),
                (lptr_t)((char*)Wlds + byteoff),
                0, 0);
        }
        __builtin_amdgcn_s_wait_asynccnt(0);
#else
        #pragma unroll
        for (int j = 0; j < 16; ++j) {
            int byteoff = j * 4096 + threadIdx.x * 16;
            *(uint4*)((char*)Wlds + byteoff) = *(const uint4*)(wsrc + byteoff);
        }
#endif
        __syncthreads();

        // ---- 8 N-tiles of width 16 out of this half ----
        if (active) {
            #pragma unroll 1
            for (int nt = 0; nt < 8; ++nt) {
                const int n0l = nt * 16;                   // col base within LDS half
                floatx8 acc = {0.f, 0.f, 0.f, 0.f, 0.f, 0.f, 0.f, 0.f};
                const bf16_t* wcol = Wlds + (size_t)(n0l + lrow) * DD + lhi * 16;
                #pragma unroll
                for (int ks = 0; ks < 8; ++ks) {
                    bf16x16 bfrag = *(const bf16x16*)(wcol + ks * 32);  // ds_load 32B
                    acc = __builtin_amdgcn_wmma_f32_16x16x32_bf16(
                        false, afrag[ks], false, bfrag, (short)0, acc, false, false);
                }
                // C layout: VGPR r -> (M = r + 8*lhi, N = lrow)
                float* crow = C + (size_t)(m0 + lhi * 8) * DD + half * 128 + n0l + lrow;
                #pragma unroll
                for (int r = 0; r < 8; ++r) crow[(size_t)r * DD] = acc[r];
            }
        }
        __syncthreads();
    }
}

// ---------------- scatter-add kernels (4 incidences per 256-thread block) ----------------

__global__ __launch_bounds__(256) void scatter_add_kernel(const float* __restrict__ in,
                                                          float* __restrict__ out,
                                                          const int* __restrict__ iidx,
                                                          const int* __restrict__ oidx, int e) {
    int ed = blockIdx.x * 4 + (threadIdx.x >> 6);          // wave-uniform edge index
    if (ed >= e) return;
    int c0 = (threadIdx.x & 63) * 4;
    int is = iidx[ed], os = oidx[ed];
    float4 v = *(const float4*)(in + (size_t)is * DD + c0);
    float* o = out + (size_t)os * DD + c0;
    atomicAdd(o + 0, v.x); atomicAdd(o + 1, v.y);
    atomicAdd(o + 2, v.z); atomicAdd(o + 3, v.w);
}

__global__ __launch_bounds__(256) void scatter_add_scaled_kernel(const float* __restrict__ in,
                                                                 const float* __restrict__ scale,
                                                                 float* __restrict__ out,
                                                                 const int* __restrict__ iidx,
                                                                 const int* __restrict__ oidx, int e) {
    int ed = blockIdx.x * 4 + (threadIdx.x >> 6);
    if (ed >= e) return;
    int c0 = (threadIdx.x & 63) * 4;
    int is = iidx[ed], os = oidx[ed];
    float s = scale[is];
    float4 v = *(const float4*)(in + (size_t)is * DD + c0);
    float* o = out + (size_t)os * DD + c0;
    atomicAdd(o + 0, s * v.x); atomicAdd(o + 1, s * v.y);
    atomicAdd(o + 2, s * v.z); atomicAdd(o + 3, s * v.w);
}

// ---------------- fused epilogues ----------------

__device__ __forceinline__ float prelu_f(float v, float a) { return (v >= 0.f) ? v : a * v; }

// h = prelu(d_inv * acc + bias); emit bf16 activations for next GEMM
__global__ __launch_bounds__(256) void post1_kernel(const float4* __restrict__ acc,
                                                    const float* __restrict__ dinv,
                                                    const float4* __restrict__ bias,
                                                    const float* __restrict__ prelu_a,
                                                    bf16x4* __restrict__ outbf, int n4) {
    int i = blockIdx.x * blockDim.x + threadIdx.x;
    if (i >= n4) return;
    int node = i >> 6;                  // 64 float4 per row
    int c4   = i & 63;
    float di = dinv[node];
    float a  = prelu_a[0];
    float4 b = bias[c4];
    float4 v = acc[i];
    float r0 = prelu_f(di * v.x + b.x, a);
    float r1 = prelu_f(di * v.y + b.y, a);
    float r2 = prelu_f(di * v.z + b.z, a);
    float r3 = prelu_f(di * v.w + b.w, a);
    bf16x4 o; o.x = (bf16_t)r0; o.y = (bf16_t)r1; o.z = (bf16_t)r2; o.w = (bf16_t)r3;
    outbf[i] = o;
}

// out = prelu(d_inv * acc + bias + x_residual)
__global__ __launch_bounds__(256) void final_kernel(const float4* __restrict__ acc,
                                                    const float* __restrict__ dinv,
                                                    const float4* __restrict__ bias,
                                                    const float4* __restrict__ xres,
                                                    const float* __restrict__ prelu_a,
                                                    float4* __restrict__ out, int n4) {
    int i = blockIdx.x * blockDim.x + threadIdx.x;
    if (i >= n4) return;
    int node = i >> 6;
    int c4   = i & 63;
    float di = dinv[node];
    float a  = prelu_a[0];
    float4 b = bias[c4];
    float4 v = acc[i];
    float4 x = xres[i];
    float4 r;
    r.x = prelu_f(di * v.x + b.x + x.x, a);
    r.y = prelu_f(di * v.y + b.y + x.y, a);
    r.z = prelu_f(di * v.z + b.z + x.z, a);
    r.w = prelu_f(di * v.w + b.w + x.w, a);
    out[i] = r;
}

// ---------------- host orchestration ----------------

extern "C" void kernel_launch(void* const* d_in, const int* in_sizes, int n_in,
                              void* d_out, int out_size, void* d_ws, size_t ws_size,
                              hipStream_t stream) {
    const float* x  = (const float*)d_in[0];   // [NN, DD]
    const float* W1 = (const float*)d_in[1];   // [DD, DD]
    const float* b1 = (const float*)d_in[2];   // [DD]
    const float* W2 = (const float*)d_in[3];   // [DD, DD]
    const float* b2 = (const float*)d_in[4];   // [DD]
    const float* pa = (const float*)d_in[5];   // scalar
    const int*   hei = (const int*)d_in[6];    // [2, EC]  row0=node(src), row1=hyperedge
    const int* src = hei;
    const int* he  = hei + EC;
    float* out = (float*)d_out;

    // workspace carve-out (256B aligned slabs)
    size_t off = 0;
    auto carve = [&](size_t bytes) -> char* {
        char* p = (char*)d_ws + off;
        off = (off + bytes + 255) & ~(size_t)255;
        return p;
    };
    float*  d_inv = (float*)carve((size_t)NN * 4);
    float*  b_inv = (float*)carve((size_t)NE * 4);
    bf16_t* Abf   = (bf16_t*)carve((size_t)NN * DD * 2);   // x_bf16, later h_bf16
    bf16_t* W1bf  = (bf16_t*)carve((size_t)DD * DD * 2);
    bf16_t* W2bf  = (bf16_t*)carve((size_t)DD * DD * 2);
    float*  bufP  = (float*)carve((size_t)NN * DD * 4);
    float*  bufQ  = (float*)carve((size_t)NN * DD * 4);
    (void)ws_size; (void)n_in; (void)in_sizes; (void)out_size;

    const int n4feat = NN * DD / 4;                 // 3,200,000
    const int gFeat  = (n4feat + 255) / 256;        // 12500
    const int gW4    = (DD * DD / 4 + 255) / 256;   // 64
    const int gDeg4  = (NN / 4 + 255) / 256;
    const int gE     = (EC + 255) / 256;
    const int gInv   = (NN + 255) / 256;
    const int gGemm  = (NN / 16 + 7) / 8;           // 391 blocks of 8 waves
    const int gScat  = (EC + 3) / 4;                // 4 edges per block

    // degrees -> inverse degrees
    zero4_kernel<<<gDeg4, 256, 0, stream>>>((float4*)d_inv, NN / 4);
    zero4_kernel<<<gDeg4, 256, 0, stream>>>((float4*)b_inv, NE / 4);
    count_deg_kernel<<<gE, 256, 0, stream>>>(src, he, d_inv, b_inv, EC);
    invert_kernel<<<gInv, 256, 0, stream>>>(d_inv, b_inv, NN);

    // bf16 copies of x, W1, W2
    cvt_f32_to_bf16x4_kernel<<<gFeat, 256, 0, stream>>>((const float4*)x, (bf16x4*)Abf, n4feat);
    cvt_f32_to_bf16x4_kernel<<<gW4, 256, 0, stream>>>((const float4*)W1, (bf16x4*)W1bf, DD * DD / 4);
    cvt_f32_to_bf16x4_kernel<<<gW4, 256, 0, stream>>>((const float4*)W2, (bf16x4*)W2bf, DD * DD / 4);

    // ---- conv1 ----
    gemm_bf16_wmma_kernel<<<gGemm, 256, 0, stream>>>(Abf, W1bf, bufP, NN);          // xw1 = x @ W1^T
    zero4_kernel<<<gFeat, 256, 0, stream>>>((float4*)bufQ, n4feat);
    scatter_add_kernel<<<gScat, 256, 0, stream>>>(bufP, bufQ, src, he, EC);         // edge_sum
    zero4_kernel<<<gFeat, 256, 0, stream>>>((float4*)bufP, n4feat);
    scatter_add_scaled_kernel<<<gScat, 256, 0, stream>>>(bufQ, b_inv, bufP, he, src, EC); // node_acc (B^-1 fused)
    post1_kernel<<<gFeat, 256, 0, stream>>>((const float4*)bufP, d_inv, (const float4*)b1,
                                            pa, (bf16x4*)Abf, n4feat);              // h = prelu(D^-1 acc + b1)

    // ---- conv2 ----
    gemm_bf16_wmma_kernel<<<gGemm, 256, 0, stream>>>(Abf, W2bf, bufQ, NN);          // xw2 = h @ W2^T
    zero4_kernel<<<gFeat, 256, 0, stream>>>((float4*)bufP, n4feat);
    scatter_add_kernel<<<gScat, 256, 0, stream>>>(bufQ, bufP, src, he, EC);         // edge_sum2
    zero4_kernel<<<gFeat, 256, 0, stream>>>((float4*)bufQ, n4feat);
    scatter_add_scaled_kernel<<<gScat, 256, 0, stream>>>(bufP, b_inv, bufQ, he, src, EC); // node_acc2
    final_kernel<<<gFeat, 256, 0, stream>>>((const float4*)bufQ, d_inv, (const float4*)b2,
                                            (const float4*)x, pa, (float4*)out, n4feat);
}
